// RNNAggregator_52819507806822
// MI455X (gfx1250) — matrix operand order repaired and checked by reading, not compiled, not run on hardware
//
#include <hip/hip_runtime.h>
#include <cstdint>
#include <cstddef>

// ---------------------------------------------------------------------------
// CDNA5 (gfx1250) implementation of the TGN GRU message aggregator.
//   Phase 1: counting sort by node id + in-segment rank sort by timestamp
//   Phase 2: convert sorted messages / transposed weights to bf16
//   Phase 3: batched GRU scan, 16 nodes per workgroup, WMMA bf16->f32.
//            Weights register-resident; hidden state in f32 REGISTERS
//            (lane-owned columns) + bf16 shadow tile in LDS for the A operand.
//            Gate activations use gfx1250 hardware transcendentals
//            (v_exp_f32 / v_rcp_f32 / v_tanh_f32).
//   Phase 4: final MLP via WMMA on bf16 hidden states
// ---------------------------------------------------------------------------

typedef __attribute__((ext_vector_type(16))) __bf16        v16bf;
typedef __attribute__((ext_vector_type(8)))  float         v8f;
typedef __attribute__((ext_vector_type(8)))  unsigned int  v8u;

#define HD 128   // hidden / message dim
#define G3 384   // 3 * HD

static __device__ __forceinline__ unsigned bf_bits(float f) {
  unsigned u = __builtin_bit_cast(unsigned, f);
  return (u + 0x7FFFu + ((u >> 16) & 1u)) >> 16;   // round-to-nearest-even
}
static __device__ __forceinline__ unsigned pack_bf2(float lo, float hi) {
  return bf_bits(lo) | (bf_bits(hi) << 16);
}
// fast sigmoid: v_exp_f32 + v_add + v_rcp_f32 (no IEEE division expansion)
static __device__ __forceinline__ float sigmoid_(float x) {
  return __builtin_amdgcn_rcpf(1.0f + __expf(-x));
}
// fast tanh: gfx1250 V_TANH_F32 when the builtin exists, else exp+rcp
static __device__ __forceinline__ float tanh_(float x) {
#if __has_builtin(__builtin_amdgcn_tanhf)
  return __builtin_amdgcn_tanhf(x);
#else
  float e = __expf(-2.0f * x);
  return 1.0f - 2.0f * __builtin_amdgcn_rcpf(1.0f + e);
#endif
}

// --------------------------- sorting kernels -------------------------------

__global__ void k_zero_i32(int* p, int n) {
  int i = blockIdx.x * blockDim.x + threadIdx.x;
  if (i < n) p[i] = 0;
}

__global__ void k_hist(const int* __restrict__ index, int* __restrict__ counts, int N) {
  int i = blockIdx.x * blockDim.x + threadIdx.x;
  if (i < N) atomicAdd(&counts[index[i]], 1);
}

// single-workgroup exclusive prefix sum over `dim` counts -> offsets[dim+1]
__global__ __launch_bounds__(1024) void k_scan(const int* __restrict__ counts,
                                               int* __restrict__ offsets,
                                               int* __restrict__ cursor, int dim) {
  __shared__ int buf[1024];
  __shared__ int carry;
  int tid = threadIdx.x;
  if (tid == 0) carry = 0;
  __syncthreads();
  for (int base = 0; base < dim; base += 1024) {
    int i = base + tid;
    int v = (i < dim) ? counts[i] : 0;
    buf[tid] = v;
    __syncthreads();
    for (int off = 1; off < 1024; off <<= 1) {
      int t = (tid >= off) ? buf[tid - off] : 0;
      __syncthreads();
      buf[tid] += t;
      __syncthreads();
    }
    if (i < dim) {
      int o = carry + buf[tid] - v;   // exclusive
      offsets[i] = o;
      cursor[i]  = o;
    }
    __syncthreads();
    if (tid == 0) carry += buf[1023];
    __syncthreads();
  }
  if (tid == 0) offsets[dim] = carry;
}

__global__ void k_scatter(const int* __restrict__ index, int* __restrict__ cursor,
                          int* __restrict__ perm0, int N) {
  int i = blockIdx.x * blockDim.x + threadIdx.x;
  if (i < N) {
    int pos = atomicAdd(&cursor[index[i]], 1);
    perm0[pos] = i;
  }
}

// within each node segment, order by (t asc, original id asc)  [stable lexsort]
__global__ void k_rank(const int* __restrict__ perm0, const int* __restrict__ index,
                       const float* __restrict__ t, const int* __restrict__ offsets,
                       int* __restrict__ permS, int N) {
  int p = blockIdx.x * blockDim.x + threadIdx.x;
  if (p >= N) return;
  int id   = perm0[p];
  int node = index[id];
  int st = offsets[node], en = offsets[node + 1];
  float ti = t[id];
  int rank = 0;
  for (int q = st; q < en; ++q) {
    int jd = perm0[q];
    float tj = t[jd];
    rank += (tj < ti) || (tj == ti && jd < id);
  }
  permS[st + rank] = id;
}

// --------------------------- bf16 conversion -------------------------------

__global__ void k_cvt_msg(const float* __restrict__ msg, const int* __restrict__ permS,
                          unsigned* __restrict__ MsgS, int N) {
  int j = blockIdx.x * blockDim.x + threadIdx.x;   // over N*64
  if (j >= N * (HD / 2)) return;
  int p  = j >> 6;
  int cu = j & 63;
  int src = permS[p];
  const float* row = msg + (size_t)src * HD + cu * 2;
  MsgS[j] = pack_bf2(row[0], row[1]);
}

__global__ void k_cvt_w(const float* __restrict__ W_ih, const float* __restrict__ W_hh,
                        const float* __restrict__ W_mlp,
                        unsigned short* __restrict__ WihT, unsigned short* __restrict__ WhhT,
                        unsigned short* __restrict__ WmlpT) {
  int i = blockIdx.x * blockDim.x + threadIdx.x;
  if (i < HD * G3) {
    int k = i / G3, n = i % G3;
    WihT[i] = (unsigned short)bf_bits(W_ih[n * HD + k]);
    WhhT[i] = (unsigned short)bf_bits(W_hh[n * HD + k]);
  }
  if (i < HD * HD) {
    int k = i / HD, n = i % HD;
    WmlpT[i] = (unsigned short)bf_bits(W_mlp[n * HD + k]);
  }
}

// --------------------------- batched GRU scan (WMMA) -----------------------
//
// One workgroup (256 thr = 8 waves) advances 16 node chains in lock-step.
// Wave w owns gate column-tiles {w, w+8, w+16}: hidden columns 16w..16w+15 of
// all three gates, so gate math is lane-local. The f32 hidden state lives in
// 8 registers per lane (that lane's column x its 8 C/D rows). LDS holds only
// a bf16 shadow of h (for the shared A operand) and the staged msg tile.

__global__ __launch_bounds__(256) void k_gru(
    const unsigned* __restrict__ MsgS,        // [N][64] packed bf16 pairs
    const unsigned short* __restrict__ WihT,  // [128][384] bf16 bits
    const unsigned short* __restrict__ WhhT,  // [128][384] bf16 bits
    const float* __restrict__ b_ih, const float* __restrict__ b_hh,
    const int* __restrict__ offsets,          // [dim+1]
    unsigned short* __restrict__ HfinB,       // [dim][128] bf16 bits
    int dim, int N) {
  __shared__ unsigned mT[16][HD / 2];   // msg tile, packed bf16 pairs
  __shared__ unsigned hbT[16][HD / 2];  // bf16 shadow of hidden tile
  __shared__ int s_start[16], s_cnt[16], s_max;

  int g    = blockIdx.x;
  int tid  = threadIdx.x;
  int lane = tid & 31;
  int w    = tid >> 5;                  // wave 0..7

  for (int i = tid; i < 16 * (HD / 2); i += 256) ((unsigned*)hbT)[i] = 0u;
  if (tid == 0) s_max = 0;
  if (tid < 16) {
    int node = g * 16 + tid;
    int st = 0, cn = 0;
    if (node < dim) { st = offsets[node]; cn = offsets[node + 1] - st; }
    s_start[tid] = st;
    s_cnt[tid]   = cn;
  }
  __syncthreads();
  if (tid < 16) atomicMax(&s_max, s_cnt[tid]);
  __syncthreads();
  int steps = s_max;

  // ---- preload register-resident B operands (contiguous k = 16h + s order) ----
  int bn  = lane & 15;                  // N within tile
  int bkh = (lane >> 4) * 16;           // K half within 32-K block
  v16bf Bih[3][4], Bhh[3][4];
  float bi[3], bh[3];
#pragma unroll
  for (int j = 0; j < 3; ++j) {         // gate: 0=r 1=z 2=n
    int col = 128 * j + 16 * w + bn;
#pragma unroll
    for (int kk = 0; kk < 4; ++kk) {
      int k0 = kk * 32 + bkh;
      v8u ui, uh;
#pragma unroll
      for (int r = 0; r < 8; ++r) {
        int k = k0 + 2 * r;
        ui[r] = (unsigned)WihT[k * G3 + col] | ((unsigned)WihT[(k + 1) * G3 + col] << 16);
        uh[r] = (unsigned)WhhT[k * G3 + col] | ((unsigned)WhhT[(k + 1) * G3 + col] << 16);
      }
      Bih[j][kk] = __builtin_bit_cast(v16bf, ui);
      Bhh[j][kk] = __builtin_bit_cast(v16bf, uh);
    }
    bi[j] = b_ih[col];
    bh[j] = b_hh[col];
  }
  float brz0 = bi[0] + bh[0];           // fused r-gate bias
  float brz1 = bi[1] + bh[1];           // fused z-gate bias

  // A-operand lane mapping (contiguous k = 16h + s, same as B)
  int am    = lane & 15;                // M (node row)
  int ahalf = (lane >> 4) * 8;          // unused; A loads use bkh
  int ccol  = 16 * w + (lane & 15);     // hidden column this lane owns
  int crow  = (lane >> 4) * 8;          // C/D row base

  float hreg[8] = {0.f, 0.f, 0.f, 0.f, 0.f, 0.f, 0.f, 0.f};
  int cntv[8];
#pragma unroll
  for (int v = 0; v < 8; ++v) cntv[v] = s_cnt[crow + v];

  for (int s = 0; s < steps; ++s) {
    // stage msg rows for step s (clamped for finished/empty chains; masked later)
    for (int i = tid; i < 16 * (HD / 2); i += 256) {
      int m  = i >> 6;
      int cu = i & 63;
      int cn = s_cnt[m];
      int srow = s_start[m] + (s < cn ? s : (cn > 0 ? cn - 1 : 0));
      if (srow >= N) srow = N - 1;
      ((unsigned*)mT)[i] = MsgS[(size_t)srow * (HD / 2) + cu];
    }
    __syncthreads();   // mT ready; previous-step hbT commits visible

    // build A operands: pure packed-uint LDS loads, zero conversion math
    v16bf Am[4], Ah[4];
#pragma unroll
    for (int kk = 0; kk < 4; ++kk) {
      v8u um, uh;
#pragma unroll
      for (int r = 0; r < 8; ++r) {
        int k = kk * 32 + bkh + 2 * r;
        um[r] = mT[am][k >> 1];
        uh[r] = hbT[am][k >> 1];
      }
      Am[kk] = __builtin_bit_cast(v16bf, um);
      Ah[kk] = __builtin_bit_cast(v16bf, uh);
    }
    __syncthreads();   // all waves captured A; safe to commit hbT below

    // gi = msg @ W_ih^T, gh = h @ W_hh^T  (24 wmma per wave per step)
    v8f aI[3] = {{}, {}, {}};
    v8f aH[3] = {{}, {}, {}};
#pragma unroll
    for (int j = 0; j < 3; ++j) {
#pragma unroll
      for (int kk = 0; kk < 4; ++kk) {
        aI[j] = __builtin_amdgcn_wmma_f32_16x16x32_bf16(
            false, Am[kk], false, Bih[j][kk], (short)0, aI[j], false, false);
        aH[j] = __builtin_amdgcn_wmma_f32_16x16x32_bf16(
            false, Ah[kk], false, Bhh[j][kk], (short)0, aH[j], false, false);
      }
    }

    // gates + hidden update (lane-local; masked commit per chain)
#pragma unroll
    for (int v = 0; v < 8; ++v) {
      float r = sigmoid_(aI[0][v] + aH[0][v] + brz0);
      float z = sigmoid_(aI[1][v] + aH[1][v] + brz1);
      float n = tanh_(aI[2][v] + bi[2] + r * (aH[2][v] + bh[2]));
      float hn = n + z * (hreg[v] - n);          // (1-z)*n + z*h
      if (s < cntv[v]) {
        hreg[v] = hn;
        ((unsigned short*)&hbT[crow + v][0])[ccol] = (unsigned short)bf_bits(hn);
      }
    }
    // loop-top barrier (after staging) orders these writes for next iteration
  }

  // write final hidden state (bf16) straight from registers
#pragma unroll
  for (int v = 0; v < 8; ++v) {
    int node = g * 16 + crow + v;
    if (node < dim) HfinB[(size_t)node * HD + ccol] = (unsigned short)bf_bits(hreg[v]);
  }
}

// --------------------------- final MLP (WMMA) ------------------------------

__global__ __launch_bounds__(256) void k_mlp(
    const unsigned* __restrict__ HfinBu,      // [dim][64] packed bf16 pairs
    const unsigned short* __restrict__ WmlpT, // [128][128] bf16 bits
    const float* __restrict__ b_mlp, const int* __restrict__ offsets,
    float* __restrict__ out, int dim) {
  __shared__ unsigned aT[16][HD / 2];
  int g = blockIdx.x, tid = threadIdx.x, lane = tid & 31, w = tid >> 5;

  for (int i = tid; i < 16 * (HD / 2); i += 256) {
    int m = i >> 6, cu = i & 63;
    int node = g * 16 + m;
    ((unsigned*)aT)[i] = (node < dim) ? HfinBu[(size_t)node * (HD / 2) + cu] : 0u;
  }
  __syncthreads();

  int bn = lane & 15, bkh = (lane >> 4) * 16;
  int am = lane & 15, ahalf = (lane >> 4) * 8;
  int col = 16 * w + bn;

  v8f acc = {};
#pragma unroll
  for (int kk = 0; kk < 4; ++kk) {
    v8u ua, ub;
#pragma unroll
    for (int r = 0; r < 8; ++r) {
      int ka = kk * 32 + bkh + 2 * r;
      ua[r] = aT[am][ka >> 1];
      int kb = kk * 32 + bkh + 2 * r;
      ub[r] = (unsigned)WmlpT[kb * HD + col] | ((unsigned)WmlpT[(kb + 1) * HD + col] << 16);
    }
    acc = __builtin_amdgcn_wmma_f32_16x16x32_bf16(
        false, __builtin_bit_cast(v16bf, ua), false, __builtin_bit_cast(v16bf, ub),
        (short)0, acc, false, false);
  }

  float bias = b_mlp[col];
  int rb = (lane >> 4) * 8;
#pragma unroll
  for (int v = 0; v < 8; ++v) {
    int node = g * 16 + rb + v;
    if (node < dim) {
      int cnt = offsets[node + 1] - offsets[node];
      out[(size_t)node * HD + col] = (cnt > 0) ? (acc[v] + bias) : 0.0f;
    }
  }
}

// --------------------------- launcher --------------------------------------

extern "C" void kernel_launch(void* const* d_in, const int* in_sizes, int n_in,
                              void* d_out, int out_size, void* d_ws, size_t ws_size,
                              hipStream_t stream) {
  const float* msg   = (const float*)d_in[0];
  const float* t     = (const float*)d_in[1];
  const int*   index = (const int*)d_in[2];
  // d_in[3] = dim_size scalar (value recovered from out_size instead)
  const float* W_ih  = (const float*)d_in[4];
  const float* W_hh  = (const float*)d_in[5];
  const float* b_ih  = (const float*)d_in[6];
  const float* b_hh  = (const float*)d_in[7];
  const float* W_mlp = (const float*)d_in[8];
  const float* b_mlp = (const float*)d_in[9];

  const int N   = in_sizes[1];                 // number of messages
  const int OUT = in_sizes[9];                 // 128
  const int dim = out_size / OUT;              // number of nodes

  uintptr_t p = (uintptr_t)d_ws;
  auto carve = [&](size_t bytes) -> void* {
    void* r = (void*)p;
    p += (bytes + 255) & ~(size_t)255;
    return r;
  };
  int*            counts  = (int*)carve((size_t)dim * 4);
  int*            offsets = (int*)carve((size_t)(dim + 1) * 4);
  int*            cursor  = (int*)carve((size_t)dim * 4);
  int*            perm0   = (int*)carve((size_t)N * 4);
  int*            permS   = (int*)carve((size_t)N * 4);
  unsigned*       MsgS    = (unsigned*)carve((size_t)N * (HD / 2) * 4);
  unsigned short* WihT    = (unsigned short*)carve((size_t)HD * G3 * 2);
  unsigned short* WhhT    = (unsigned short*)carve((size_t)HD * G3 * 2);
  unsigned short* WmlpT   = (unsigned short*)carve((size_t)HD * HD * 2);
  unsigned short* HfinB   = (unsigned short*)carve((size_t)dim * HD * 2);
  (void)ws_size; (void)n_in;

  const int B = 256;
  // Phase 1: sort by (node, t)
  k_zero_i32<<<(dim + B - 1) / B, B, 0, stream>>>(counts, dim);
  k_hist<<<(N + B - 1) / B, B, 0, stream>>>(index, counts, N);
  k_scan<<<1, 1024, 0, stream>>>(counts, offsets, cursor, dim);
  k_scatter<<<(N + B - 1) / B, B, 0, stream>>>(index, cursor, perm0, N);
  k_rank<<<(N + B - 1) / B, B, 0, stream>>>(perm0, index, t, offsets, permS, N);

  // Phase 2: bf16 conversion
  int cvtN = N * (HD / 2);
  k_cvt_msg<<<(cvtN + B - 1) / B, B, 0, stream>>>(msg, permS, MsgS, N);
  k_cvt_w<<<(HD * G3 + B - 1) / B, B, 0, stream>>>(W_ih, W_hh, W_mlp, WihT, WhhT, WmlpT);

  // Phase 3: batched GRU scan (16 chains per workgroup)
  k_gru<<<(dim + 15) / 16, 256, 0, stream>>>(MsgS, WihT, WhhT, b_ih, b_hh,
                                             offsets, HfinB, dim, N);

  // Phase 4: output MLP
  k_mlp<<<(dim + 15) / 16, 256, 0, stream>>>((const unsigned*)HfinB, WmlpT, b_mlp,
                                             offsets, (float*)d_out, dim);
}
